// PitchEnergyPredictor_4483945857370
// MI455X (gfx1250) — matrix operand order repaired and checked by reading, not compiled, not run on hardware
//
#include <hip/hip_runtime.h>

typedef __attribute__((ext_vector_type(16))) __bf16 v16bf;
typedef __attribute__((ext_vector_type(8)))  __bf16 v8bf;
typedef __attribute__((ext_vector_type(8)))  float  v8f;

#define BB   4
#define TT   512
#define FFR  2048
#define SDIM 128
#define DDIM 512
#define CCH  640
#define NHD  8
#define DHD  80
#define HID  256
#define GH   1024   /* 4*HID */
#define C2   1280   /* 2*CCH */

__device__ __forceinline__ float sigmf(float x){ return 1.f/(1.f+expf(-x)); }

/* ---------------- generic bf16 WMMA GEMM, double-buffered ----------------
 * C[b][m][n] (+)= sum_k A[b][m+rowOff][k] * B[b][k][n]  (+ bias[n] when acc==0)
 * A rows outside [0,Arows) read as zero (used for k3 conv shifts).
 * block = 256 thr = 8 waves; tile 128x64; each wave owns 32x32 = 4 wmma tiles.
 * Pipeline: global loads for tile k+1 issue before the WMMAs of tile k;
 * LDS is double-buffered so only ONE barrier per K-step is needed.
 * B tile is stored TRANSPOSED in LDS so fragments are ds_load_b128s.      */
__global__ __launch_bounds__(256) void k_gemm(
    const __bf16* __restrict__ A, long sAb, int lda, int Arows, int rowOff,
    const __bf16* __restrict__ Bm, long sBb, int ldb,
    float* __restrict__ Cm, long sCb, int ldc,
    const float* __restrict__ bias, int M, int N, int K, int accFlag)
{
  __shared__ __bf16 As [2][128][32];   /* [buf][m][k] */
  __shared__ __bf16 BsT[2][ 64][32];   /* [buf][n][k] transposed */
  const int bz = blockIdx.z;
  const __bf16* Ab = A + (long)bz * sAb;
  const __bf16* Bb = Bm + (long)bz * sBb;
  float* Cb = Cm + (long)bz * sCb;
  const int m0 = blockIdx.y * 128;
  const int n0 = blockIdx.x * 64;
  const int tid  = threadIdx.x;
  const int wave = tid >> 5, lane = tid & 31;
  const int wm = wave >> 1, wn = wave & 1;
  const int hlf = lane >> 4;
  const int l16 = lane & 15;
  const int kb   = hlf * 8;
  const int koff = hlf * 16;
  /* per-thread staging coordinates */
  const int ar0 = tid >> 3, akc = (tid & 7) << 2;        /* A: rows ar0 + 32*it */
  const int br0 = tid >> 4, bnc = (tid & 15) << 2;       /* B: rows br0 + 16*it */
  const bool fullTile = (m0 + rowOff >= 0) && (m0 + 127 + rowOff < Arows);

  uint2 aReg[4], bReg[2];
  auto loadTile = [&](int k0) {
    if (fullTile) {
      #pragma unroll
      for (int it = 0; it < 4; ++it) {
        int gr = m0 + (ar0 + it*32) + rowOff;
        aReg[it] = *(const uint2*)(Ab + (long)gr * lda + (k0 + akc));
      }
    } else {
      #pragma unroll
      for (int it = 0; it < 4; ++it) {
        int gr = m0 + (ar0 + it*32) + rowOff;
        uint2 v; v.x = 0u; v.y = 0u;
        if (gr >= 0 && gr < Arows) v = *(const uint2*)(Ab + (long)gr * lda + (k0 + akc));
        aReg[it] = v;
      }
    }
    #pragma unroll
    for (int it = 0; it < 2; ++it) {
      int r = br0 + it*16;
      bReg[it] = *(const uint2*)(Bb + (long)(k0 + r) * ldb + (n0 + bnc));
    }
    if (k0 + 32 < K && tid < 128) {                 /* prefetch hint one more tile out */
      int gr = m0 + tid + rowOff;
      if (gr >= 0 && gr < Arows)
        __builtin_prefetch(&Ab[(long)gr * lda + k0 + 32], 0, 1);
    }
  };
  auto storeTile = [&](int p) {
    #pragma unroll
    for (int it = 0; it < 4; ++it)
      *(uint2*)(&As[p][ar0 + it*32][akc]) = aReg[it];
    #pragma unroll
    for (int it = 0; it < 2; ++it) {
      int r = br0 + it*16;
      const __bf16* v4 = (const __bf16*)&bReg[it];
      #pragma unroll
      for (int j = 0; j < 4; ++j) BsT[p][bnc + j][r] = v4[j];
    }
  };

  v8f acc00 = {0.f,0.f,0.f,0.f,0.f,0.f,0.f,0.f};
  v8f acc01 = acc00, acc10 = acc00, acc11 = acc00;

  loadTile(0);
  storeTile(0);
  __syncthreads();
  int p = 0;
  for (int k0 = 0; k0 < K; k0 += 32) {
    const bool hasNext = (k0 + 32) < K;
    if (hasNext) loadTile(k0 + 32);                 /* overlap with WMMAs below */
    /* ---- fragments from buffer p: contiguous vector LDS loads ---- */
    const __bf16* ap0 = &As[p][wm*32 + l16][0];
    const __bf16* ap1 = &As[p][wm*32 + 16 + l16][0];
    v8bf a0lo = *(const v8bf*)(ap0 + koff);
    v8bf a0hi = *(const v8bf*)(ap0 + koff + 8);
    v8bf a1lo = *(const v8bf*)(ap1 + koff);
    v8bf a1hi = *(const v8bf*)(ap1 + koff + 8);
    v16bf af0 = __builtin_shufflevector(a0lo, a0hi, 0,1,2,3,4,5,6,7,8,9,10,11,12,13,14,15);
    v16bf af1 = __builtin_shufflevector(a1lo, a1hi, 0,1,2,3,4,5,6,7,8,9,10,11,12,13,14,15);
    v16bf b0 = *(const v16bf*)(&BsT[p][wn*32 + l16][koff]);
    v16bf b1 = *(const v16bf*)(&BsT[p][wn*32 + 16 + l16][koff]);
    acc00 = __builtin_amdgcn_wmma_f32_16x16x32_bf16(false, af0, false, b0, (short)0, acc00, false, false);
    acc01 = __builtin_amdgcn_wmma_f32_16x16x32_bf16(false, af0, false, b1, (short)0, acc01, false, false);
    acc10 = __builtin_amdgcn_wmma_f32_16x16x32_bf16(false, af1, false, b0, (short)0, acc10, false, false);
    acc11 = __builtin_amdgcn_wmma_f32_16x16x32_bf16(false, af1, false, b1, (short)0, acc11, false, false);
    if (hasNext) storeTile(p ^ 1);                  /* other buffer: no race with reads */
    __syncthreads();                                /* single barrier per K-step */
    p ^= 1;
  }
  /* ---- epilogue ---- */
  const int colA = n0 + wn*32 + l16, colB = colA + 16;
  #pragma unroll
  for (int sub = 0; sub < 2; ++sub) {
    int rowbase = m0 + wm*32 + sub*16 + hlf*8;
    v8f aA = sub ? acc10 : acc00;
    v8f aB = sub ? acc11 : acc01;
    #pragma unroll
    for (int r = 0; r < 8; ++r) {
      int row = rowbase + r;
      float v0 = aA[r], v1 = aB[r];
      long i0 = (long)row * ldc + colA;
      long i1 = (long)row * ldc + colB;
      if (accFlag) { Cb[i0] += v0; Cb[i1] += v1; }
      else {
        if (bias) { v0 += bias[colA]; v1 += bias[colB]; }
        Cb[i0] = v0; Cb[i1] = v1;
      }
    }
  }
}

/* strided f32 -> bf16 convert (handles transposes / k-slices) */
__global__ void k_convertT(const float* __restrict__ src, __bf16* __restrict__ dst,
                           int R, int Cc, long sr, long sc, long sb, long db, long off)
{
  long i = (long)blockIdx.x * blockDim.x + threadIdx.x;
  long tot = (long)R * Cc;
  if (i >= tot) return;
  int b = blockIdx.z;
  long r = i / Cc, c = i % Cc;
  dst[(long)b*db + i] = (__bf16)src[(long)b*sb + r*sr + c*sc + off];
}

/* out[b,o] = bias[o] + style[b,:] . w[o,:]   (w is [O,S]) */
__global__ void k_style_mod(const float* __restrict__ style, const float* __restrict__ w,
                            const float* __restrict__ bias, float* __restrict__ out, int O)
{
  int i = blockIdx.x * blockDim.x + threadIdx.x;
  if (i >= BB * O) return;
  int b = i / O, o = i % O;
  const float* s  = style + b * SDIM;
  const float* wr = w + (long)o * SDIM;
  float acc = bias[o];
  for (int k = 0; k < SDIM; ++k) acc += s[k] * wr[k];
  out[i] = acc;
}

/* x0 = concat(text_enc^T, style) * valid ; write f32 + bf16 */
__global__ void k_build_x0(const float* __restrict__ te, const float* __restrict__ style,
                           const int* __restrict__ lens, float* __restrict__ xf, __bf16* __restrict__ xb)
{
  long i = (long)blockIdx.x * blockDim.x + threadIdx.x;
  if (i >= (long)BB*TT*CCH) return;
  int c = (int)(i % CCH);
  long bt = i / CCH;
  int t = (int)(bt % TT), b = (int)(bt / TT);
  float v = (c < DDIM) ? te[((long)b*DDIM + c)*TT + t] : style[b*SDIM + (c - DDIM)];
  if (t >= lens[b]) v = 0.f;
  xf[i] = v; xb[i] = (__bf16)v;
}

/* one workgroup per (b,dir): sequential LSTM recurrence, h/c in LDS */
__global__ __launch_bounds__(1024) void k_lstm(const float* __restrict__ gates,
    const float* __restrict__ Whh, float* __restrict__ hcat)
{
  int b = blockIdx.x & 3;
  int d = blockIdx.x >> 2;
  const float* gsrc = gates + (long)d*BB*TT*GH + (long)b*TT*GH;
  const float* W = Whh + (long)d*GH*HID;
  __shared__ float h[HID], c[HID], g[GH];
  int tid = threadIdx.x;
  if (tid < HID) { h[tid] = 0.f; c[tid] = 0.f; }
  for (int step = 0; step < TT; ++step) {
    int t = d ? (TT-1-step) : step;
    __syncthreads();
    float acc = gsrc[(long)t*GH + tid];
    const float* wr = W + (long)tid*HID;
    #pragma unroll 4
    for (int j = 0; j < HID; ++j) acc += h[j] * wr[j];
    g[tid] = acc;
    __syncthreads();
    if (tid < HID) {
      float gi = g[tid], gf = g[HID+tid], gg = g[2*HID+tid], go = g[3*HID+tid];
      float cc = sigmf(gf)*c[tid] + sigmf(gi)*tanhf(gg);
      c[tid] = cc;
      float hh = sigmf(go)*tanhf(cc);
      h[tid] = hh;
      hcat[((long)b*TT + t)*DDIM + d*HID + tid] = hh;
    }
  }
}

/* AdaLayerNorm over 512 + style concat + mask -> x f32/bf16 (one block / row) */
__global__ __launch_bounds__(256) void k_adaln_pe(const float* __restrict__ hcat, const float* __restrict__ gb,
    const float* __restrict__ style, const int* __restrict__ lens,
    float* __restrict__ xf, __bf16* __restrict__ xb)
{
  int row = blockIdx.x;                   /* b*TT + t */
  int b = row / TT, t = row % TT;
  const float* xr = hcat + (long)row * DDIM;
  __shared__ float red[256];
  float s = 0.f, sq = 0.f;
  for (int c = threadIdx.x; c < DDIM; c += 256) { float v = xr[c]; s += v; sq += v*v; }
  red[threadIdx.x] = s; __syncthreads();
  for (int st = 128; st; st >>= 1) { if (threadIdx.x < st) red[threadIdx.x] += red[threadIdx.x+st]; __syncthreads(); }
  float mean = red[0] / DDIM; __syncthreads();
  red[threadIdx.x] = sq; __syncthreads();
  for (int st = 128; st; st >>= 1) { if (threadIdx.x < st) red[threadIdx.x] += red[threadIdx.x+st]; __syncthreads(); }
  float var = red[0] / DDIM - mean*mean;
  float rstd = rsqrtf(var + 1e-5f);
  float valid = (t < lens[b]) ? 1.f : 0.f;
  for (int c = threadIdx.x; c < CCH; c += 256) {
    float v;
    if (c < DDIM) v = (1.f + gb[b*GH + c]) * ((xr[c]-mean)*rstd) + gb[b*GH + DDIM + c];
    else          v = style[b*SDIM + (c - DDIM)];
    v *= valid;
    long o = (long)row*CCH + c;
    xf[o] = v; xb[o] = (__bf16)v;
  }
}

/* AdaLayerNorm over C=640 -> bf16 only (q / k paths) */
__global__ __launch_bounds__(256) void k_adaln_c(const float* __restrict__ x, const float* __restrict__ gb,
    __bf16* __restrict__ outb, int R)
{
  int row = blockIdx.x;                    /* b*R + r */
  int b = row / R;
  const float* xr = x + (long)row * CCH;
  __shared__ float red[256];
  float s = 0.f, sq = 0.f;
  for (int c = threadIdx.x; c < CCH; c += 256) { float v = xr[c]; s += v; sq += v*v; }
  red[threadIdx.x] = s; __syncthreads();
  for (int st = 128; st; st >>= 1) { if (threadIdx.x < st) red[threadIdx.x] += red[threadIdx.x+st]; __syncthreads(); }
  float mean = red[0] / CCH; __syncthreads();
  red[threadIdx.x] = sq; __syncthreads();
  for (int st = 128; st; st >>= 1) { if (threadIdx.x < st) red[threadIdx.x] += red[threadIdx.x+st]; __syncthreads(); }
  float var = red[0] / CCH - mean*mean;
  float rstd = rsqrtf(var + 1e-5f);
  const float* g = gb + (long)b * C2;
  for (int c = threadIdx.x; c < CCH; c += 256) {
    float y = (1.f + g[c]) * ((xr[c]-mean)*rstd) + g[CCH + c];
    outb[(long)row*CCH + c] = (__bf16)y;
  }
}

__global__ void k_tau(const float* __restrict__ align, int* __restrict__ tau)
{
  int i = blockIdx.x * blockDim.x + threadIdx.x;
  if (i >= BB * FFR) return;
  int b = i / FFR, f = i % FFR;
  const float* p = align + (long)b*TT*FFR + f;
  float mx = p[0]; int am = 0;
  for (int t = 1; t < TT; ++t) { float v = p[(long)t*FFR]; if (v > mx) { mx = v; am = t; } }
  tau[i] = am;
}

__global__ void k_vsum(const float* __restrict__ v, float* __restrict__ out)
{
  int i = blockIdx.x * blockDim.x + threadIdx.x;
  if (i >= BB * CCH) return;
  int b = i / CCH, c = i % CCH;
  float s = 0.f;
  for (int t = 0; t < TT; ++t) s += v[((long)b*TT + t)*CCH + c];
  out[i] = s;
}

/* band attention: softmax over the 11-wide window (everything else underflows to 0) */
__global__ void k_attn(const float* __restrict__ qh, const float* __restrict__ kh,
                       const float* __restrict__ vh, const int* __restrict__ tau,
                       const int* __restrict__ lens, const float* __restrict__ vsum,
                       float* __restrict__ out)
{
  long i = (long)blockIdx.x * blockDim.x + threadIdx.x;
  if (i >= (long)BB*FFR*NHD) return;
  int h = (int)(i % NHD);
  long bf = i / NHD;
  int f = (int)(bf % FFR), b = (int)(bf / FFR);
  const float* q = qh + ((long)b*FFR + f)*CCH + h*DHD;
  int tv = tau[b*FFR + f], len = lens[b];
  float sc[11]; float mx = -3.4e38f; int any = 0;
  for (int w = 0; w < 11; ++w) {
    int t = tv - 5 + w;
    float s = -3.4e38f;
    if (t >= 0 && t < TT && t < len) {
      const float* k = kh + ((long)b*TT + t)*CCH + h*DHD;
      float d = 0.f;
      for (int e = 0; e < DHD; ++e) d += q[e]*k[e];
      s = d * 0.11180339887498949f;   /* 1/sqrt(80) */
      any = 1; if (s > mx) mx = s;
    }
    sc[w] = s;
  }
  float* o = out + ((long)b*FFR + f)*CCH + h*DHD;
  if (any) {
    float p[11]; float den = 0.f;
    for (int w = 0; w < 11; ++w) { float e = (sc[w] > -1e37f) ? expf(sc[w]-mx) : 0.f; p[w] = e; den += e; }
    float inv = 1.f / den;
    for (int e = 0; e < DHD; ++e) {
      float a = 0.f;
      for (int w = 0; w < 11; ++w) {
        if (p[w] > 0.f) { int t = tv - 5 + w; a += p[w] * vh[((long)b*TT + t)*CCH + h*DHD + e]; }
      }
      o[e] = a * inv;
    }
  } else {        /* all -1e4: reference softmax -> uniform over all T */
    for (int e = 0; e < DHD; ++e) o[e] = vsum[b*CCH + h*DHD + e] * (1.f/512.f);
  }
}

__global__ void k_dwconv_silu(const float* __restrict__ x, const float* __restrict__ w,
                              const float* __restrict__ bias, __bf16* __restrict__ outb)
{
  long i = (long)blockIdx.x * blockDim.x + threadIdx.x;
  if (i >= (long)BB*FFR*CCH) return;
  int c = (int)(i % CCH);
  long bf = i / CCH;
  int f = (int)(bf % FFR), b = (int)(bf / FFR);
  float acc = bias[c];
  for (int j = 0; j < 5; ++j) {
    int ff = f + j - 2;
    if (ff >= 0 && ff < FFR) acc += x[((long)b*FFR + ff)*CCH + c] * w[c*5 + j];
  }
  float y = acc * sigmf(acc);
  outb[i] = (__bf16)y;
}

__global__ void k_addscale2(const float* __restrict__ a, const float* __restrict__ b, float scale,
                            float* __restrict__ o1, float* __restrict__ o2, long n)
{
  long i = (long)blockIdx.x * blockDim.x + threadIdx.x;
  if (i >= n) return;
  float v = (a[i] + b[i]) * scale;
  o1[i] = v;
  if (o2) o2[i] = v;
}

__global__ void k_adain_stats(const float* __restrict__ x, float* __restrict__ stats)
{
  int i = blockIdx.x * blockDim.x + threadIdx.x;
  if (i >= BB * CCH) return;
  int b = i / CCH, c = i % CCH;
  const float* p = x + (long)b*FFR*CCH + c;
  float s = 0.f, sq = 0.f;
  for (int f = 0; f < FFR; ++f) { float v = p[(long)f*CCH]; s += v; sq += v*v; }
  float mu = s / FFR;
  float var = sq / FFR - mu*mu;
  stats[i*2]   = mu;
  stats[i*2+1] = rsqrtf(var + 1e-5f);
}

__global__ void k_adain_apply(const float* __restrict__ x, const float* __restrict__ stats,
                              const float* __restrict__ gb, __bf16* __restrict__ outb)
{
  long i = (long)blockIdx.x * blockDim.x + threadIdx.x;
  if (i >= (long)BB*FFR*CCH) return;
  int c = (int)(i % CCH);
  int b = (int)(i / ((long)FFR*CCH));
  int si = b*CCH + c;
  float xn = (x[i] - stats[si*2]) * stats[si*2+1];
  float y = (1.f + gb[b*C2 + c]) * xn + gb[b*C2 + CCH + c];
  y = (y > 0.f) ? y : 0.2f * y;
  outb[i] = (__bf16)y;
}

__global__ void k_proj1(const float* __restrict__ x, const float* __restrict__ w,
                        const float* __restrict__ bias, float* __restrict__ out)
{
  int i = blockIdx.x * blockDim.x + threadIdx.x;
  if (i >= BB * FFR) return;
  const float* xr = x + (long)i * CCH;
  float acc = bias[0];
  for (int c = 0; c < CCH; ++c) acc += xr[c] * w[c];
  out[i] = acc;
}

extern "C" void kernel_launch(void* const* d_in, const int* in_sizes, int n_in,
                              void* d_out, int out_size, void* d_ws, size_t ws_size,
                              hipStream_t stream)
{
  (void)in_sizes; (void)n_in; (void)out_size;
  const float* te    = (const float*)d_in[0];
  const int*   lens  = (const int*  )d_in[1];
  const float* align = (const float*)d_in[2];
  const float* style = (const float*)d_in[3];
  const float* peWih = (const float*)d_in[4];
  const float* peWhh = (const float*)d_in[5];
  const float* peB   = (const float*)d_in[6];
  const float* peLnW = (const float*)d_in[7];
  const float* peLnB = (const float*)d_in[8];
  const float* qnW = (const float*)d_in[9];  const float* qnB = (const float*)d_in[10];
  const float* knW = (const float*)d_in[11]; const float* knB = (const float*)d_in[12];
  const float* wqp = (const float*)d_in[13]; const float* wqb = (const float*)d_in[14];
  const float* wkp = (const float*)d_in[15]; const float* wkb = (const float*)d_in[16];
  const float* wvp = (const float*)d_in[17]; const float* wvb = (const float*)d_in[18];
  const float* wop = (const float*)d_in[19]; const float* wob = (const float*)d_in[20];
  const float* dww = (const float*)d_in[21]; const float* dwbp= (const float*)d_in[22];
  const float* pww = (const float*)d_in[23]; const float* pwbp= (const float*)d_in[24];
  const float* fc1w[2] = { (const float*)d_in[25], (const float*)d_in[35] };
  const float* fc1b[2] = { (const float*)d_in[26], (const float*)d_in[36] };
  const float* cv1w[2] = { (const float*)d_in[27], (const float*)d_in[37] };
  const float* cv1b[2] = { (const float*)d_in[28], (const float*)d_in[38] };
  const float* fc2w[2] = { (const float*)d_in[29], (const float*)d_in[39] };
  const float* fc2b[2] = { (const float*)d_in[30], (const float*)d_in[40] };
  const float* cv2w[2] = { (const float*)d_in[31], (const float*)d_in[41] };
  const float* cv2b[2] = { (const float*)d_in[32], (const float*)d_in[42] };
  const float* prjw[2] = { (const float*)d_in[33], (const float*)d_in[43] };
  const float* prjb[2] = { (const float*)d_in[34], (const float*)d_in[44] };
  float* out = (float*)d_out;

  /* ---- workspace arena ---- */
  char* base = (char*)d_ws;
  size_t cur = 0;
  auto alloc = [&](size_t bytes) -> void* {
    void* r = base + cur;
    cur = (cur + bytes + 255) & ~(size_t)255;
    return r;
  };
  float*  xf    = (float*) alloc((size_t)BB*TT*CCH*4);
  __bf16* xb    = (__bf16*)alloc((size_t)BB*TT*CCH*2);
  float*  hcat  = (float*) alloc((size_t)BB*TT*DDIM*4);
  float*  gates = (float*) alloc((size_t)2*BB*TT*GH*4);
  __bf16* wT    = (__bf16*)alloc((size_t)CCH*GH*2);
  float*  gb    = (float*) alloc((size_t)BB*C2*4);
  __bf16* alT   = (__bf16*)alloc((size_t)BB*FFR*TT*2);
  float*  baseT = (float*) alloc((size_t)BB*FFR*CCH*4);
  float*  t1f   = (float*) alloc((size_t)BB*FFR*CCH*4);
  float*  t2f   = (float*) alloc((size_t)BB*FFR*CCH*4);
  __bf16* t1b   = (__bf16*)alloc((size_t)BB*FFR*CCH*2);
  __bf16* ksb   = (__bf16*)alloc((size_t)BB*TT*CCH*2);
  float*  khT   = (float*) alloc((size_t)BB*TT*CCH*4);
  float*  vhT   = (float*) alloc((size_t)BB*TT*CCH*4);
  int*    tau   = (int*)   alloc((size_t)BB*FFR*4);
  float*  vsum  = (float*) alloc((size_t)BB*CCH*4);
  float*  f0T   = (float*) alloc((size_t)BB*FFR*CCH*4);
  float*  enT   = (float*) alloc((size_t)BB*FFR*CCH*4);
  float*  h2    = (float*) alloc((size_t)BB*FFR*CCH*4);
  float*  stats = (float*) alloc((size_t)BB*CCH*2*4);
  if (cur > ws_size) return;

  auto blks = [](long n) { return dim3((unsigned)((n + 255) / 256)); };
  auto gemm = [&](const __bf16* A, long sAb, int lda, int Arows, int rowOff,
                  const __bf16* Bm, long sBb, int ldb,
                  float* Cm, long sCb, int ldc, const float* bias,
                  int M, int N, int K, int batches, int acc) {
    dim3 g(N/64, M/128, batches);
    k_gemm<<<g, 256, 0, stream>>>(A, sAb, lda, Arows, rowOff, Bm, sBb, ldb,
                                  Cm, sCb, ldc, bias, M, N, K, acc);
  };
  auto convT = [&](const float* src, __bf16* dst, int R, int Cc, long sr, long sc,
                   long sb, long db, int batches, long off) {
    dim3 g((unsigned)(((long)R*Cc + 255)/256), 1, batches);
    k_convertT<<<g, 256, 0, stream>>>(src, dst, R, Cc, sr, sc, sb, db, off);
  };
  auto smod = [&](const float* w, const float* b, int O, float* o) {
    k_style_mod<<<blks((long)BB*O), 256, 0, stream>>>(style, w, b, o, O);
  };

  /* ---- ProsodyEncoder ---- */
  k_build_x0<<<blks((long)BB*TT*CCH), 256, 0, stream>>>(te, style, lens, xf, xb);
  for (int l = 0; l < 3; ++l) {
    for (int d = 0; d < 2; ++d) {
      convT(peWih + (size_t)(l*2+d)*GH*CCH, wT, CCH, GH, 1, CCH, 0, 0, 1, 0);
      gemm(xb, 0, CCH, BB*TT, 0, wT, 0, GH,
           gates + (size_t)d*BB*TT*GH, 0, GH,
           peB + (size_t)(l*2+d)*GH, BB*TT, GH, CCH, 1, 0);
    }
    k_lstm<<<dim3(BB*2), 1024, 0, stream>>>(gates, peWhh + (size_t)l*2*GH*HID, hcat);
    smod(peLnW + (size_t)l*GH*SDIM, peLnB + (size_t)l*GH, GH, gb);
    k_adaln_pe<<<dim3(BB*TT), 256, 0, stream>>>(hcat, gb, style, lens, xf, xb);
  }

  /* ---- cross ---- */
  convT(align, alT, FFR, TT, 1, FFR, (long)TT*FFR, (long)FFR*TT, BB, 0);
  gemm(alT, (long)FFR*TT, TT, FFR, 0, xb, (long)TT*CCH, CCH,
       baseT, (long)FFR*CCH, CCH, nullptr, FFR, CCH, TT, BB, 0);

  smod(qnW, qnB, C2, gb);
  k_adaln_c<<<dim3(BB*FFR), 256, 0, stream>>>(baseT, gb, t1b, FFR);
  convT(wqp, wT, CCH, CCH, 1, CCH, 0, 0, 1, 0);
  gemm(t1b, (long)FFR*CCH, CCH, FFR, 0, wT, 0, CCH,
       t1f, (long)FFR*CCH, CCH, wqb, FFR, CCH, CCH, BB, 0);

  smod(knW, knB, C2, gb);
  k_adaln_c<<<dim3(BB*TT), 256, 0, stream>>>(xf, gb, ksb, TT);
  convT(wkp, wT, CCH, CCH, 1, CCH, 0, 0, 1, 0);
  gemm(ksb, (long)TT*CCH, CCH, TT, 0, wT, 0, CCH,
       khT, (long)TT*CCH, CCH, wkb, TT, CCH, CCH, BB, 0);
  convT(wvp, wT, CCH, CCH, 1, CCH, 0, 0, 1, 0);
  gemm(ksb, (long)TT*CCH, CCH, TT, 0, wT, 0, CCH,
       vhT, (long)TT*CCH, CCH, wvb, TT, CCH, CCH, BB, 0);

  k_tau<<<blks((long)BB*FFR), 256, 0, stream>>>(align, tau);
  k_vsum<<<blks((long)BB*CCH), 256, 0, stream>>>(vhT, vsum);
  k_attn<<<blks((long)BB*FFR*NHD), 256, 0, stream>>>(t1f, khT, vhT, tau, lens, vsum, t2f);

  convT(t2f, t1b, BB*FFR, CCH, CCH, 1, 0, 0, 1, 0);
  convT(wop, wT, CCH, CCH, 1, CCH, 0, 0, 1, 0);
  gemm(t1b, 0, CCH, BB*FFR, 0, wT, 0, CCH, t2f, 0, CCH, wob, BB*FFR, CCH, CCH, 1, 0);

  k_dwconv_silu<<<blks((long)BB*FFR*CCH), 256, 0, stream>>>(t2f, dww, dwbp, t1b);
  convT(pww, wT, CCH, CCH, 1, CCH, 0, 0, 1, 0);
  gemm(t1b, 0, CCH, BB*FFR, 0, wT, 0, CCH, t2f, 0, CCH, pwbp, BB*FFR, CCH, CCH, 1, 0);

  k_addscale2<<<blks((long)BB*FFR*CCH), 256, 0, stream>>>(
      baseT, t2f, 0.70710678118654752f, f0T, enT, (long)BB*FFR*CCH);

  /* ---- F0 / N heads: 3 AdaIN residual blocks + 1x1 proj ---- */
  for (int hd = 0; hd < 2; ++hd) {
    float* curb = (hd == 0) ? f0T : enT;
    for (int l = 0; l < 3; ++l) {
      smod(fc1w[hd] + (size_t)l*C2*SDIM, fc1b[hd] + (size_t)l*C2, C2, gb);
      k_adain_stats<<<blks((long)BB*CCH), 256, 0, stream>>>(curb, stats);
      k_adain_apply<<<blks((long)BB*FFR*CCH), 256, 0, stream>>>(curb, stats, gb, t1b);
      const float* cw1 = cv1w[hd] + (size_t)l*CCH*CCH*3;
      for (int dk = 0; dk < 3; ++dk) {
        convT(cw1, wT, CCH, CCH, 3, (long)3*CCH, 0, 0, 1, dk);
        gemm(t1b, (long)FFR*CCH, CCH, FFR, dk-1, wT, 0, CCH,
             h2, (long)FFR*CCH, CCH,
             (dk == 0) ? (cv1b[hd] + (size_t)l*CCH) : nullptr,
             FFR, CCH, CCH, BB, (dk == 0) ? 0 : 1);
      }
      smod(fc2w[hd] + (size_t)l*C2*SDIM, fc2b[hd] + (size_t)l*C2, C2, gb);
      k_adain_stats<<<blks((long)BB*CCH), 256, 0, stream>>>(h2, stats);
      k_adain_apply<<<blks((long)BB*FFR*CCH), 256, 0, stream>>>(h2, stats, gb, t1b);
      const float* cw2 = cv2w[hd] + (size_t)l*CCH*CCH*3;
      for (int dk = 0; dk < 3; ++dk) {
        convT(cw2, wT, CCH, CCH, 3, (long)3*CCH, 0, 0, 1, dk);
        gemm(t1b, (long)FFR*CCH, CCH, FFR, dk-1, wT, 0, CCH,
             h2, (long)FFR*CCH, CCH,
             (dk == 0) ? (cv2b[hd] + (size_t)l*CCH) : nullptr,
             FFR, CCH, CCH, BB, (dk == 0) ? 0 : 1);
      }
      k_addscale2<<<blks((long)BB*FFR*CCH), 256, 0, stream>>>(
          curb, h2, 0.70710678118654752f, curb, nullptr, (long)BB*FFR*CCH);
    }
    k_proj1<<<blks((long)BB*FFR), 256, 0, stream>>>(curb, prjw[hd], prjb[hd],
                                                    out + (size_t)hd*BB*FFR);
  }
}
